// SwitchMoE_62079457296767
// MI455X (gfx1250) — matrix operand order repaired and checked by reading, not compiled, not run on hardware
//
#include <hip/hip_runtime.h>
#include <hip/hip_bf16.h>
#include <math.h>

// ---------------------------------------------------------------------------
// Switch-MoE (top-1) for MI455X / gfx1250, wave32 + WMMA bf16.
//   gate -> route (pad-to-32 per expert) -> grouped GEMM1 (gelu) -> GEMM2.
// GEMM math: bf16 operands (converted on the fly from fp32 HBM), fp32
// accumulation via v_wmma_f32_16x16x32_bf16.  M-tile = 32 rows so each
// staged 32x128 B chunk feeds 2 WMMAs per wave (amortizes staging VALU).
// ---------------------------------------------------------------------------

#define DEV __device__ __forceinline__

constexpr int TT = 4096;                 // tokens (B*T)
constexpr int D_ = 1024;                 // model dim
constexpr int F_ = 4096;                 // hidden dim
constexpr int E_ = 8;                    // experts
constexpr int MTILE = 32;                // rows per GEMM tile
constexpr int MAX_PAD   = TT + E_ * MTILE;   // 4352 padded-sorted rows
constexpr int MAX_TILES = MAX_PAD / MTILE;   // 136 row tiles

typedef __attribute__((ext_vector_type(4)))  unsigned u32x4;
typedef __attribute__((ext_vector_type(4)))  float    f32x4;
typedef __attribute__((ext_vector_type(2)))  float    f32x2;
typedef __attribute__((ext_vector_type(8)))  float    v8f;
typedef __attribute__((ext_vector_type(16))) __bf16   v16bf;

// pack two fp32 -> two bf16 (cheap round-toward-nearest via +0x8000)
DEV unsigned pk_bf16(float a, float b) {
  unsigned ua = __builtin_bit_cast(unsigned, a) + 0x8000u;
  unsigned ub = __builtin_bit_cast(unsigned, b) + 0x8000u;
  return (ua >> 16) | (ub & 0xFFFF0000u);
}

// ---------------------------------------------------------------------------
// Gate: one wave per token. red = x[t] @ w_red (16), cosine vs wg rows,
// softmax over 8, top-1 idx + score.
// ---------------------------------------------------------------------------
__global__ void moe_gate_kernel(const float* __restrict__ x,
                                const float* __restrict__ w_red,
                                const float* __restrict__ wg,
                                int* __restrict__ idx_out,
                                float* __restrict__ score_out) {
  const int lane = threadIdx.x & 31;
  const int t    = blockIdx.x * 8 + (threadIdx.x >> 5);  // 8 waves / block

  float acc[16];
#pragma unroll
  for (int j = 0; j < 16; ++j) acc[j] = 0.f;

  const float* xr = x + (size_t)t * D_;
  for (int k = lane; k < D_; k += 32) {
    float xv = xr[k];
    const f32x4* wr = (const f32x4*)(w_red + k * 16);
#pragma unroll
    for (int q = 0; q < 4; ++q) {
      f32x4 w4 = wr[q];
      acc[4 * q + 0] += xv * w4.x;
      acc[4 * q + 1] += xv * w4.y;
      acc[4 * q + 2] += xv * w4.z;
      acc[4 * q + 3] += xv * w4.w;
    }
  }
#pragma unroll
  for (int off = 16; off > 0; off >>= 1)
#pragma unroll
    for (int j = 0; j < 16; ++j) acc[j] += __shfl_xor(acc[j], off, 32);

  if (lane == 0) {
    float lg[E_];
#pragma unroll
    for (int e = 0; e < E_; ++e) {
      const float* w = wg + e * 16;
      float nn = 0.f, dot = 0.f;
#pragma unroll
      for (int j = 0; j < 16; ++j) { nn += w[j] * w[j]; dot += w[j] * acc[j]; }
      lg[e] = dot / fmaxf(sqrtf(nn), 1e-4f);
    }
    float mx = lg[0]; int bi = 0;
#pragma unroll
    for (int e = 1; e < E_; ++e) if (lg[e] > mx) { mx = lg[e]; bi = e; }
    float s = 0.f;
#pragma unroll
    for (int e = 0; e < E_; ++e) s += expf(lg[e] - mx);
    idx_out[t]   = bi;
    score_out[t] = 1.f / s;  // exp(mx-mx)/sum
  }
}

// ---------------------------------------------------------------------------
// Routing: histogram, pad each expert segment to multiple of 32, scatter
// token ids into sorted order, emit tile->expert map (launch-constant grid).
// ---------------------------------------------------------------------------
__global__ void moe_route_kernel(const int* __restrict__ idx,
                                 int* __restrict__ sorted_tok,
                                 int* __restrict__ tile_e) {
  __shared__ int cnt[E_];
  __shared__ int seg[E_ + 1];
  __shared__ int cur[E_];
  const int t = threadIdx.x;

  if (t < E_) cnt[t] = 0;
  __syncthreads();
  for (int i = t; i < TT; i += blockDim.x) atomicAdd(&cnt[idx[i]], 1);
  __syncthreads();
  if (t == 0) {
    seg[0] = 0;
    for (int e = 0; e < E_; ++e) seg[e + 1] = seg[e] + ((cnt[e] + 31) & ~31);
    for (int e = 0; e < E_; ++e) cur[e] = seg[e];
  }
  __syncthreads();
  const int ntiles = seg[E_] / MTILE;
  for (int i = t; i < MAX_PAD; i += blockDim.x) sorted_tok[i] = -1;
  for (int i = t; i < MAX_TILES; i += blockDim.x) {
    int e = -1;
    if (i < ntiles) {
      for (int q = 0; q < E_; ++q)
        if (i * MTILE >= seg[q] && i * MTILE < seg[q + 1]) e = q;
    }
    tile_e[i] = e;
  }
  __syncthreads();
  for (int i = t; i < TT; i += blockDim.x) {
    int e = idx[i];
    int p = atomicAdd(&cur[e], 1);
    sorted_tok[p] = i;
  }
}

// ---------------------------------------------------------------------------
// Shared GEMM helpers.  LDS layouts (uint packs 2 bf16 along K):
//   sA[row][kp]  row-stride 20  (32 rows  x 16 k-pairs)   -> conflict-free
//   sB[n][kp]    row-stride 20  (128 cols x 16 k-pairs)   -> conflict-free
// A-frag per ISA: lanes 0-15 K{0..7,16..23}, lanes16-31 K{8..15,24..31}
// B-frag (matches A): lanes 0-15 K{0..7,16..23}, lanes16-31 K{8..15,24..31}
// ---------------------------------------------------------------------------
DEV void stage_chunk_B(const float* __restrict__ Bp, int ldb,
                       unsigned* __restrict__ sB, int t) {
  const int kp = t >> 4;  // 0..15 -> k = 2*kp, 2*kp+1
  const float* r0 = Bp + (size_t)(2 * kp) * ldb;
  const float* r1 = r0 + ldb;
#pragma unroll
  for (int q = 0; q < 2; ++q) {
    int n4 = 4 * (t & 15) + 64 * q;
    f32x4 a = *(const f32x4*)(r0 + n4);
    f32x4 b = *(const f32x4*)(r1 + n4);
    sB[(n4 + 0) * 20 + kp] = pk_bf16(a.x, b.x);
    sB[(n4 + 1) * 20 + kp] = pk_bf16(a.y, b.y);
    sB[(n4 + 2) * 20 + kp] = pk_bf16(a.z, b.z);
    sB[(n4 + 3) * 20 + kp] = pk_bf16(a.w, b.w);
  }
}

DEV v16bf load_fragA(const unsigned* __restrict__ sA, int lane, int half16) {
  const int m = (lane & 15) + half16 * 16, half = lane >> 4;
  const unsigned* p = sA + m * 20 + half * 4;
  union { u32x4 q[2]; v16bf v; } u;
  u.q[0] = *(const u32x4*)p;        // K 0..7   (or 8..15)
  u.q[1] = *(const u32x4*)(p + 8);  // K 16..23 (or 24..31)
  return u.v;
}

DEV v16bf load_fragB(const unsigned* __restrict__ sB, int lane, int wv) {
  const int n = wv * 16 + (lane & 15), half = lane >> 4;
  const unsigned* p = sB + n * 20 + half * 4;
  union { u32x4 q[2]; v16bf v; } u;
  u.q[0] = *(const u32x4*)p;        // K 0..7   (or 8..15)
  u.q[1] = *(const u32x4*)(p + 8);  // K 16..23 (or 24..31)
  return u.v;
}

// ---------------------------------------------------------------------------
// FFN layer 1:  h[row, F] = gelu( x[tok] @ W1[e] + b1[e] )   (bf16 out)
// grid (MAX_TILES, F/128), block 256 (8 waves, 2x 16x16 C tiles per wave)
// ---------------------------------------------------------------------------
__global__ void moe_ffn1_kernel(const float* __restrict__ x,
                                const float* __restrict__ w1,
                                const float* __restrict__ b1,
                                const int* __restrict__ sorted_tok,
                                const int* __restrict__ tile_e,
                                unsigned short* __restrict__ h) {
  __shared__ unsigned sB[128 * 20];
  __shared__ unsigned sA[MTILE * 20];
  __shared__ unsigned short sH[MTILE][136];
  __shared__ int sToks[MTILE];

  const int tile = blockIdx.x;
  const int e = tile_e[tile];
  if (e < 0) return;
  const int t = threadIdx.x;
  if (t < MTILE) sToks[t] = sorted_tok[tile * MTILE + t];
  __syncthreads();

  const int wv = t >> 5, lane = t & 31;
  const int ar0 = t >> 4, ar1 = ar0 + 16, akp = t & 15;
  const int tk0 = sToks[ar0], tk1 = sToks[ar1];
  const float* xA0 = (tk0 >= 0) ? (x + (size_t)tk0 * D_ + 2 * akp) : nullptr;
  const float* xA1 = (tk1 >= 0) ? (x + (size_t)tk1 * D_ + 2 * akp) : nullptr;
  const float* Bbase = w1 + (size_t)e * D_ * F_ + (size_t)blockIdx.y * 128;

  v8f c0 = {0.f, 0.f, 0.f, 0.f, 0.f, 0.f, 0.f, 0.f};
  v8f c1 = c0;
  for (int ks = 0; ks < D_ / 32; ++ks) {
    __syncthreads();
    f32x2 fa0 = {0.f, 0.f}, fa1 = {0.f, 0.f};
    if (xA0) fa0 = *(const f32x2*)(xA0 + ks * 32);
    if (xA1) fa1 = *(const f32x2*)(xA1 + ks * 32);
    sA[ar0 * 20 + akp] = pk_bf16(fa0.x, fa0.y);
    sA[ar1 * 20 + akp] = pk_bf16(fa1.x, fa1.y);
    stage_chunk_B(Bbase + (size_t)(ks * 32) * F_, F_, sB, t);
    __syncthreads();
    v16bf bv = load_fragB(sB, lane, wv);
    v16bf a0 = load_fragA(sA, lane, 0);
    v16bf a1 = load_fragA(sA, lane, 1);
    c0 = __builtin_amdgcn_wmma_f32_16x16x32_bf16(false, a0, false, bv,
                                                 (short)0, c0, false, false);
    c1 = __builtin_amdgcn_wmma_f32_16x16x32_bf16(false, a1, false, bv,
                                                 (short)0, c1, false, false);
  }

  // epilogue: bias + exact gelu, bf16, coalesced store via LDS
  const int col = wv * 16 + (lane & 15);
  const int half = lane >> 4;
  const float bias = b1[e * F_ + blockIdx.y * 128 + col];
#pragma unroll
  for (int r = 0; r < 8; ++r) {
    int m = r + 8 * half;
    float v0 = c0[r] + bias;
    v0 = 0.5f * v0 * (1.f + erff(v0 * 0.7071067811865475f));
    unsigned u0 = __builtin_bit_cast(unsigned, v0) + 0x8000u;
    sH[m][col] = (unsigned short)(u0 >> 16);
    float v1 = c1[r] + bias;
    v1 = 0.5f * v1 * (1.f + erff(v1 * 0.7071067811865475f));
    unsigned u1 = __builtin_bit_cast(unsigned, v1) + 0x8000u;
    sH[m + 16][col] = (unsigned short)(u1 >> 16);
  }
  __syncthreads();
  const int row = t >> 4, seg = t & 15;
  u32x4 o0 = *(const u32x4*)&sH[row][seg * 8];
  u32x4 o1 = *(const u32x4*)&sH[row + 16][seg * 8];
  unsigned short* hp =
      h + (size_t)(tile * MTILE + row) * F_ + blockIdx.y * 128 + seg * 8;
  *(u32x4*)hp = o0;
  *(u32x4*)(hp + (size_t)16 * F_) = o1;
}

// ---------------------------------------------------------------------------
// FFN layer 2:  out[tok] = (h[row] @ W2[e] + b2[e]) * score[tok]
// grid (MAX_TILES, D/128), block 256
// ---------------------------------------------------------------------------
__global__ void moe_ffn2_kernel(const unsigned short* __restrict__ h,
                                const float* __restrict__ w2,
                                const float* __restrict__ b2,
                                const int* __restrict__ sorted_tok,
                                const int* __restrict__ tile_e,
                                const float* __restrict__ score,
                                float* __restrict__ out) {
  __shared__ unsigned sB[128 * 20];
  __shared__ unsigned sA[MTILE * 20];
  __shared__ float sY[MTILE][132];
  __shared__ int sToks[MTILE];
  __shared__ float sScs[MTILE];

  const int tile = blockIdx.x;
  const int e = tile_e[tile];
  if (e < 0) return;
  const int t = threadIdx.x;
  if (t < MTILE) {
    int tok = sorted_tok[tile * MTILE + t];
    sToks[t] = tok;
    sScs[t] = (tok >= 0) ? score[tok] : 0.f;
  }
  __syncthreads();

  const int wv = t >> 5, lane = t & 31;
  const int ar0 = t >> 4, ar1 = ar0 + 16, akp = t & 15;
  const unsigned short* hA0 =
      h + (size_t)(tile * MTILE + ar0) * F_ + 2 * akp;
  const unsigned short* hA1 = hA0 + (size_t)16 * F_;
  const float* Bbase = w2 + (size_t)e * F_ * D_ + (size_t)blockIdx.y * 128;

  v8f c0 = {0.f, 0.f, 0.f, 0.f, 0.f, 0.f, 0.f, 0.f};
  v8f c1 = c0;
  for (int ks = 0; ks < F_ / 32; ++ks) {
    __syncthreads();
    sA[ar0 * 20 + akp] = *(const unsigned*)(hA0 + ks * 32);  // already bf16
    sA[ar1 * 20 + akp] = *(const unsigned*)(hA1 + ks * 32);
    stage_chunk_B(Bbase + (size_t)(ks * 32) * D_, D_, sB, t);
    __syncthreads();
    v16bf bv = load_fragB(sB, lane, wv);
    v16bf a0 = load_fragA(sA, lane, 0);
    v16bf a1 = load_fragA(sA, lane, 1);
    c0 = __builtin_amdgcn_wmma_f32_16x16x32_bf16(false, a0, false, bv,
                                                 (short)0, c0, false, false);
    c1 = __builtin_amdgcn_wmma_f32_16x16x32_bf16(false, a1, false, bv,
                                                 (short)0, c1, false, false);
  }

  const int col = wv * 16 + (lane & 15);
  const int half = lane >> 4;
  const float bias = b2[e * D_ + blockIdx.y * 128 + col];
#pragma unroll
  for (int r = 0; r < 8; ++r) {
    int m = r + 8 * half;
    sY[m][col]      = (c0[r] + bias) * sScs[m];
    sY[m + 16][col] = (c1[r] + bias) * sScs[m + 16];
  }
  __syncthreads();
  const int row = t >> 4, seg = t & 15;
#pragma unroll
  for (int p = 0; p < 2; ++p) {
    const int rr = row + 16 * p;
    const int tok = sToks[rr];
    if (tok >= 0) {
      f32x4 y0 = *(const f32x4*)&sY[rr][seg * 8];
      f32x4 y1 = *(const f32x4*)&sY[rr][seg * 8 + 4];
      float* op = out + (size_t)tok * D_ + blockIdx.y * 128 + seg * 8;
      *(f32x4*)op = y0;
      *(f32x4*)(op + 4) = y1;
    }
  }
}

// ---------------------------------------------------------------------------
// Launch. Workspace layout (bytes):
//   [0)       idx        int[4096]      (16384)
//   [16384)   score      float[4096]    (16384)
//   [32768)   sorted_tok int[4352]      (17408)
//   [50176)   tile_e     int[136]       (544)
//   [65536)   h          bf16[4352*4096] (35,651,584)   total ~35.7 MB
// ---------------------------------------------------------------------------
extern "C" void kernel_launch(void* const* d_in, const int* in_sizes, int n_in,
                              void* d_out, int out_size, void* d_ws, size_t ws_size,
                              hipStream_t stream) {
  (void)in_sizes; (void)n_in; (void)out_size; (void)ws_size;
  const float* x     = (const float*)d_in[0];
  const float* w_red = (const float*)d_in[1];
  const float* wg    = (const float*)d_in[2];
  const float* w1    = (const float*)d_in[3];
  const float* b1    = (const float*)d_in[4];
  const float* w2    = (const float*)d_in[5];
  const float* b2    = (const float*)d_in[6];
  float* out = (float*)d_out;

  char* ws = (char*)d_ws;
  int*            idx    = (int*)(ws + 0);
  float*          score  = (float*)(ws + 16384);
  int*            sorted = (int*)(ws + 32768);
  int*            tile_e = (int*)(ws + 50176);
  unsigned short* h      = (unsigned short*)(ws + 65536);

  moe_gate_kernel<<<TT / 8, 256, 0, stream>>>(x, w_red, wg, idx, score);
  moe_route_kernel<<<1, 256, 0, stream>>>(idx, sorted, tile_e);
  moe_ffn1_kernel<<<dim3(MAX_TILES, F_ / 128), 256, 0, stream>>>(
      x, w1, b1, sorted, tile_e, h);
  moe_ffn2_kernel<<<dim3(MAX_TILES, D_ / 128), 256, 0, stream>>>(
      h, w2, b2, sorted, tile_e, score, out);
}
